// AlphaRNN_4234837754463
// MI455X (gfx1250) — compile-verified, matrix operand order checked by reading, not run on hardware
//
#include <hip/hip_runtime.h>
#include <math.h>

// AlphaRNN (2-layer) for MI455X / gfx1250, wave32 + bf16 WMMA + TDM.
//
// Pipeline (all in place inside d_out's y1 region; zero workspace):
//   1) proj_gemm<K=128>:  xp0 = x @ W_ih0^T + b_ih0            (parallel WMMA GEMM)
//   2) alpha_scan:        y0  = scan(xp0, W_hh0, b_hh0, a0)    (in-place), h0 -> tail
//   3) proj_gemm<K=256>:  xp1 = y0 @ W_ih1^T + b_ih1           (in-place over y0)
//   4) alpha_scan:        y1  = scan(xp1, W_hh1, b_hh1, a1)    (in-place), h1 -> tail
//
// Scan: 8 WGs x 16 batch rows; 16 wave32s/WG, one 16-col N-tile each.
//  - W_hh register-resident as bf16 B-fragments (64 VGPRs/wave, reused 512 steps)
//  - h as bf16 A-fragments in LDS + fp32 in accumulator-layout registers
//  - xp[:,t,:] tiles streamed by the Tensor Data Mover into double-buffered LDS,
//    one step ahead, so the only global read overlaps the WMMA chain
//  - branch-free tanh via v_exp_f32 + v_rcp_f32 (no EXEC divergence per step)

#define B_ 128
#define S_ 512
#define I_ 128
#define H_ 256
static constexpr int BSH = B_ * S_ * H_;   // 16,777,216 floats
static constexpr int BH  = B_ * H_;        // 32,768 floats

typedef __attribute__((ext_vector_type(16))) __bf16 v16bf;
typedef __attribute__((ext_vector_type(8)))  float  v8f;
typedef __attribute__((ext_vector_type(4)))  unsigned int v4u;
typedef __attribute__((ext_vector_type(8)))  int    v8i;
typedef __attribute__((ext_vector_type(4)))  int    v4i;

union Frag {
    v16bf v;
    uint4 q[2];
};

// tanh without branches: v_exp_f32 + v_rcp_f32 + bfi(copysign).
__device__ __forceinline__ float fast_tanh(float x) {
    const float ax = fabsf(x);
    const float e  = __expf(-2.0f * ax);                  // v_exp_f32
    const float r  = __builtin_amdgcn_rcpf(1.0f + e);     // v_rcp_f32
    return copysignf((1.0f - e) * r, x);
}

// Load 8 consecutive fp32, convert to bf16, place in fragment elements e0..e0+7.
// Fragment element map (16-bit A/B 16x32 layout): element e of lane L covers
// k = kbase + (e/8)*16 + (L/16)*8 + (e%8)  -> 8 consecutive k per half.
__device__ __forceinline__ void pack8(v16bf& f, int e0, const float* p) {
    float4 a = ((const float4*)p)[0];
    float4 b = ((const float4*)p)[1];
    f[e0 + 0] = (__bf16)a.x; f[e0 + 1] = (__bf16)a.y;
    f[e0 + 2] = (__bf16)a.z; f[e0 + 3] = (__bf16)a.w;
    f[e0 + 4] = (__bf16)b.x; f[e0 + 5] = (__bf16)b.y;
    f[e0 + 6] = (__bf16)b.z; f[e0 + 7] = (__bf16)b.w;
}

// Out[m, 0:256] = A[m, 0:K] @ W[0:256, 0:K]^T + bias.  One wave per 16-row M tile.
// A and Out may alias row-for-row (layer-1 in-place projection): all A fragments
// are register-resident before any store, and waves own disjoint rows.
template <int K>
__global__ void proj_gemm_kernel(const float* A, const float* __restrict__ W,
                                 const float* __restrict__ bias, float* Out) {
    constexpr int KC = K / 32;
    const int lane = threadIdx.x & 31;
    const int wave = threadIdx.x >> 5;          // 0..3
    const int half = lane >> 4;                 // 0/1
    const int l16  = lane & 15;
    const int mtile = blockIdx.x * 4 + wave;
    const int mbase = mtile * 16;

    v16bf afrag[KC];
    const float* rowp = A + (size_t)(mbase + l16) * K;
#pragma unroll
    for (int kc = 0; kc < KC; ++kc) {
        pack8(afrag[kc], 0, rowp + kc * 32 + half * 8);
        pack8(afrag[kc], 8, rowp + kc * 32 + 16 + half * 8);
    }

    for (int nt = 0; nt < 16; ++nt) {
        v8f acc = {};
        const float* wrow = W + (size_t)(nt * 16 + l16) * K;   // B[k][n] = W[n][k]
#pragma unroll
        for (int kc = 0; kc < KC; ++kc) {
            v16bf bfrag;
            pack8(bfrag, 0, wrow + kc * 32 + half * 8);
            pack8(bfrag, 8, wrow + kc * 32 + 16 + half * 8);
            acc = __builtin_amdgcn_wmma_f32_16x16x32_bf16(
                false, afrag[kc], false, bfrag, (short)0, acc, false, false);
        }
        const float bv = bias[nt * 16 + l16];
#pragma unroll
        for (int r = 0; r < 8; ++r) {
            Out[(size_t)(mbase + r + 8 * half) * H_ + nt * 16 + l16] = acc[r] + bv;
        }
    }
}

// LDS layout for the scan kernel.
struct __align__(16) Smem {
    __bf16 h[16][264];        // 8448 B, padded pitch vs banks (16B-aligned rows)
    float  xp[2][16][256];    // 2 x 16 KB TDM staging (double buffer)
};

// Issue a TDM 2D tile load: 16 rows x 256 f32, row stride = S_*H_ elements,
// from global tile base `ga` into LDS byte address `lds_byte`.
// D# group0: count=1, lds_addr, global_addr[56:0], type=2 ("image").
// D# group1: data_size=2 (4B), tensor_dim0=256, tensor_dim1=16,
//            tile_dim0=256, tile_dim1=16, tensor_dim0_stride=131072 elems.
__device__ __forceinline__ void tdm_load_tile(unsigned long long ga,
                                              unsigned int lds_byte) {
    v4u g0;
    g0[0] = 1u;                                   // count=1, user descriptor
    g0[1] = lds_byte;                             // lds_addr (bytes)
    g0[2] = (unsigned int)ga;                     // global_addr[31:0]
    g0[3] = (unsigned int)(ga >> 32) | (2u << 30);// global_addr[56:32] | type=2
    v8i g1;
    g1[0] = 0x00020000;                           // workgroup_mask=0, data_size=2 (4B)
    g1[1] = (int)(256u << 16);                    // tensor_dim0 = 256 (bits 79:48 lo)
    g1[2] = (int)(16u << 16);                     // tensor_dim1 = 16  (bits 111:80 lo)
    g1[3] = (int)(256u << 16);                    // tile_dim0 = 256   (bits 127:112)
    g1[4] = 16;                                   // tile_dim1 = 16, tile_dim2 = 0
    g1[5] = (int)(unsigned)(S_ * H_);             // tensor_dim0_stride = 131072 elems
    g1[6] = 0;                                    // stride hi / dim1_stride lo
    g1[7] = 0;
    v4i z4 = {0, 0, 0, 0};
    v8i z8 = {0, 0, 0, 0, 0, 0, 0, 0};
    // amdgpu-toolchain (clang-23) 6-arg form: (g0, g1, g2, g3, g4, cpol)
    __builtin_amdgcn_tensor_load_to_lds(g0, g1, z4, z4, z8, 0);
}

// In-place scan: XY holds xp on entry; each [:,t,:] tile is replaced by y[:,t,:].
__global__ void __launch_bounds__(512, 1)
alpha_scan_kernel(float* XY, const float* __restrict__ Whh,
                  const float* __restrict__ bhh, const float* __restrict__ alpha,
                  float* __restrict__ hfinal) {
    __shared__ Smem sm;
    const int lane = threadIdx.x & 31;
    const int wave = threadIdx.x >> 5;          // 0..15 -> N tile
    const int half = lane >> 4;
    const int l16  = lane & 15;
    const int bbase = blockIdx.x * 16;          // batch tile
    const int ncol  = wave * 16 + l16;

    const float a   = 1.0f / (1.0f + __expf(-alpha[0]));
    const float oma = 1.0f - a;
    const float bv  = bhh[ncol];

    // W_hh B-fragments, register resident for all 512 steps (64 VGPRs).
    v16bf wfrag[8];
    const float* wrow = Whh + (size_t)ncol * H_;
#pragma unroll
    for (int kc = 0; kc < 8; ++kc) {
        pack8(wfrag[kc], 0, wrow + kc * 32 + half * 8);
        pack8(wfrag[kc], 8, wrow + kc * 32 + 16 + half * 8);
    }

    // h = 0
    for (int i = threadIdx.x; i < 16 * 264; i += 512) (&sm.h[0][0])[i] = (__bf16)0.0f;
    float hprev[8];
#pragma unroll
    for (int r = 0; r < 8; ++r) hprev[r] = 0.0f;

    // LDS byte offsets of the two xp staging buffers (workgroup-relative).
    const unsigned int lds_xp0 = (unsigned int)(uintptr_t)(&sm.xp[0][0][0]);
    const unsigned int lds_xp1 = (unsigned int)(uintptr_t)(&sm.xp[1][0][0]);
    const unsigned long long gbase =
        (unsigned long long)(uintptr_t)(XY + (size_t)bbase * S_ * H_);

    // Prologue: DMA the t=0 tile into buffer 0.
    if (wave == 0) {
        tdm_load_tile(gbase, lds_xp0);
        __builtin_amdgcn_s_wait_tensorcnt(0);
    }
    __syncthreads();

    for (int t = 0; t < S_; ++t) {
        // Kick the DMA for t+1 into the other buffer; overlaps the WMMA chain.
        if (wave == 0 && t + 1 < S_) {
            tdm_load_tile(gbase + (unsigned long long)(t + 1) * H_ * 4,
                          ((t + 1) & 1) ? lds_xp1 : lds_xp0);
        }

        // A fragments of h from LDS: two ds_load_b128 per K chunk.
        Frag af[8];
#pragma unroll
        for (int kc = 0; kc < 8; ++kc) {
            af[kc].q[0] = *(const uint4*)&sm.h[l16][kc * 32 + half * 8];
            af[kc].q[1] = *(const uint4*)&sm.h[l16][kc * 32 + 16 + half * 8];
        }

        v8f acc = {};
#pragma unroll
        for (int kc = 0; kc < 8; ++kc) {
            acc = __builtin_amdgcn_wmma_f32_16x16x32_bf16(
                false, af[kc].v, false, wfrag[kc], (short)0, acc, false, false);
        }
        __syncthreads();   // all h reads done before republish

        const float* xprow = &sm.xp[t & 1][0][0];
#pragma unroll
        for (int r = 0; r < 8; ++r) {
            const int m = r + 8 * half;
            const float xpv = xprow[m * 256 + ncol];
            const float hn  = a * hprev[r] + oma * fast_tanh(xpv + acc[r] + bv);
            hprev[r] = hn;
            XY[((size_t)(bbase + m) * S_ + t) * H_ + ncol] = hn;  // y over xp, in place
            sm.h[m][ncol] = (__bf16)hn;                           // publish h
        }

        if (wave == 0) __builtin_amdgcn_s_wait_tensorcnt(0);  // t+1 tile landed
        __syncthreads();   // h + next xp visible to all waves
    }

#pragma unroll
    for (int r = 0; r < 8; ++r)
        hfinal[(size_t)(bbase + r + 8 * half) * H_ + ncol] = hprev[r];
}

extern "C" void kernel_launch(void* const* d_in, const int* in_sizes, int n_in,
                              void* d_out, int out_size, void* d_ws, size_t ws_size,
                              hipStream_t stream) {
    (void)in_sizes; (void)n_in; (void)out_size; (void)d_ws; (void)ws_size;
    const float* x    = (const float*)d_in[0];
    const float* Wih0 = (const float*)d_in[1];
    const float* bih0 = (const float*)d_in[2];
    const float* Whh0 = (const float*)d_in[3];
    const float* bhh0 = (const float*)d_in[4];
    const float* al0  = (const float*)d_in[5];
    const float* Wih1 = (const float*)d_in[6];
    const float* bih1 = (const float*)d_in[7];
    const float* Whh1 = (const float*)d_in[8];
    const float* bhh1 = (const float*)d_in[9];
    const float* al1  = (const float*)d_in[10];

    float* y  = (float*)d_out;      // [B,S,H] staging + final y1
    float* h0 = y + BSH;
    float* h1 = h0 + BH;

    const dim3 gblk(128);                 // 4 waves, one 16-row M tile each
    const dim3 ggrd((B_ * S_) / 64);      // 65536 rows / (4*16) = 1024 blocks

    proj_gemm_kernel<I_><<<ggrd, gblk, 0, stream>>>(x, Wih0, bih0, y);
    alpha_scan_kernel<<<dim3(B_ / 16), dim3(512), 0, stream>>>(y, Whh0, bhh0, al0, h0);
    proj_gemm_kernel<H_><<<ggrd, gblk, 0, stream>>>(y, Wih1, bih1, y);   // in place
    alpha_scan_kernel<<<dim3(B_ / 16), dim3(512), 0, stream>>>(y, Whh1, bhh1, al1, h1);
}
